// CrossAttention_86062554677405
// MI455X (gfx1250) — compile-verified, matrix operand order checked by reading, not run on hardware
//
#include <hip/hip_runtime.h>
#include <hip/hip_bf16.h>

typedef __bf16 bf16;
typedef __attribute__((ext_vector_type(16))) __bf16 bfx16;
typedef __attribute__((ext_vector_type(8)))  __bf16 bfx8;
typedef __attribute__((ext_vector_type(8)))  float  fx8;

__device__ __forceinline__ bf16 f2bf(float f) {
    unsigned u = __builtin_bit_cast(unsigned, f);
    unsigned r = u + 0x7FFFu + ((u >> 16) & 1u);
    unsigned short h = (unsigned short)(r >> 16);
    return __builtin_bit_cast(bf16, h);
}

// ---------------------------------------------------------------------------
// Elementwise f32 -> bf16 cast (vectorized x4)
// ---------------------------------------------------------------------------
__global__ void ca_cast_f32_bf16(const float* __restrict__ in,
                                 bf16* __restrict__ out, long long n) {
    long long i = ((long long)blockIdx.x * blockDim.x + threadIdx.x) * 4;
    if (i + 3 < n) {
        float4 v = *(const float4*)(in + i);
        out[i + 0] = f2bf(v.x);
        out[i + 1] = f2bf(v.y);
        out[i + 2] = f2bf(v.z);
        out[i + 3] = f2bf(v.w);
    }
}

// Transpose + cast: in [R,C] f32 row-major -> out [C,R] bf16 row-major
__global__ void ca_transpose_cast(const float* __restrict__ in,
                                  bf16* __restrict__ out, int R, int C) {
    long long idx = (long long)blockIdx.x * blockDim.x + threadIdx.x;
    if (idx < (long long)R * C) {
        int rr = (int)(idx / C);
        int cc = (int)(idx % C);
        out[(long long)cc * R + rr] = f2bf(in[idx]);
    }
}

// ---------------------------------------------------------------------------
// CDNA5 async-copy helpers (ASYNCcnt-tracked DMA into LDS, no VGPR roundtrip)
// ---------------------------------------------------------------------------
__device__ __forceinline__ void async_b128_to_lds(unsigned lds_byte,
                                                  const void* gaddr) {
    asm volatile("global_load_async_to_lds_b128 %0, %1, off"
                 :: "v"(lds_byte),
                    "v"((unsigned long long)(uintptr_t)gaddr)
                 : "memory");
}
__device__ __forceinline__ void wait_async0() {
    asm volatile("s_wait_asynccnt 0x0" ::: "memory");
}

// ---------------------------------------------------------------------------
// Generic batched bf16 WMMA GEMM:
//   C[M,Ntot] = A[M,K] (row-major) * Bt[Ntot,K]^T (row-major) + bias
// Block tile 128x128x64, 256 threads = 8 waves, each wave 64x32.
// Double-buffered LDS filled by global_load_async_to_lds_b128.
// ---------------------------------------------------------------------------
#define BM 128
#define BN 128
#define BK 64
#define LDP 72   // padded LDS row stride (elems): 144B -> conflict-free, 16B aligned

__device__ __forceinline__ bfx16 frag_from_lds(const bf16* __restrict__ p) {
    // CDNA5 16-bit A/B layout: per lane, elems 0-7 = K[half*8 .. +7],
    // elems 8-15 = K[16+half*8 .. +7]  (p already offset by half*8)
    bfx8 lo = *(const bfx8*)(p);
    bfx8 hi = *(const bfx8*)(p + 16);
    bfx16 f;
#pragma unroll
    for (int i = 0; i < 8; ++i) { f[i] = lo[i]; f[i + 8] = hi[i]; }
    return f;
}

// Issue one 128x64 bf16 tile of A and B into LDS (async). 256 threads,
// 4 x b128 per thread per matrix.
__device__ __forceinline__ void issue_tile(const bf16* __restrict__ gA, int lda,
                                           const bf16* __restrict__ gB, int ldb,
                                           unsigned ldsA, unsigned ldsB, int tid) {
#pragma unroll
    for (int t = 0; t < 4; ++t) {
        int v   = tid + t * 256;        // 0..1023
        int row = v >> 3;               // 0..127
        int cv  = v & 7;                // 0..7 (8 bf16 each)
        unsigned loff = (unsigned)(row * (LDP * 2) + cv * 16);
        async_b128_to_lds(ldsA + loff, gA + (long long)row * lda + cv * 8);
        async_b128_to_lds(ldsB + loff, gB + (long long)row * ldb + cv * 8);
    }
}

__global__ __launch_bounds__(256)
void ca_gemm_bf16_wmma(const bf16* __restrict__ A,  long long strA, int lda,
                       const bf16* __restrict__ Bt, long long strB, int ldb,
                       const float* __restrict__ bias,
                       float* __restrict__ Cf, bf16* __restrict__ Cb,
                       long long strC, int ldc,
                       const float* __restrict__ resid, long long strR,
                       int storeTrans, int MperBatch, long long tbStride, int ldct,
                       int M, int Ntot, int K)
{
    __shared__ __align__(16) bf16 sA[2][BM * LDP];
    __shared__ __align__(16) bf16 sB[2][BN * LDP];

    const int tid  = threadIdx.x;
    const int lane = tid & 31;
    const int wave = tid >> 5;
    const int r    = lane & 15;
    const int half = lane >> 4;
    const int mW   = (wave >> 2) * 64;  // 2 wave-rows
    const int nW   = (wave & 3) * 32;   // 4 wave-cols
    const int blockM = blockIdx.y * BM;
    const int blockN = blockIdx.x * BN;
    const int bz     = blockIdx.z;

    const bf16* gA0 = A  + (long long)bz * strA + (long long)blockM * lda;
    const bf16* gB0 = Bt + (long long)bz * strB + (long long)blockN * ldb;

    unsigned ldsA[2], ldsB[2];
    ldsA[0] = (unsigned)(uintptr_t)&sA[0][0];
    ldsA[1] = (unsigned)(uintptr_t)&sA[1][0];
    ldsB[0] = (unsigned)(uintptr_t)&sB[0][0];
    ldsB[1] = (unsigned)(uintptr_t)&sB[1][0];

    fx8 acc[4][2] = {};

    const int nTiles = K / BK;
    issue_tile(gA0, lda, gB0, ldb, ldsA[0], ldsB[0], tid);

    for (int it = 0; it < nTiles; ++it) {
        const int cur = it & 1;
        wait_async0();        // our async writes to buffer `cur` are done
        __syncthreads();      // everyone's writes done / prior reads drained

        if (it + 1 < nTiles)  // prefill other buffer while computing
            issue_tile(gA0 + (it + 1) * BK, lda, gB0 + (it + 1) * BK, ldb,
                       ldsA[cur ^ 1], ldsB[cur ^ 1], tid);

        const bf16* cA = &sA[cur][0];
        const bf16* cB = &sB[cur][0];
#pragma unroll
        for (int ks = 0; ks < 2; ++ks) {
            const int kc = ks * 32 + half * 8;
            bfx16 bf_[2];
#pragma unroll
            for (int ni = 0; ni < 2; ++ni)
                bf_[ni] = frag_from_lds(&cB[(nW + 16 * ni + r) * LDP + kc]);
#pragma unroll
            for (int mi = 0; mi < 4; ++mi) {
                bfx16 af = frag_from_lds(&cA[(mW + 16 * mi + r) * LDP + kc]);
#pragma unroll
                for (int ni = 0; ni < 2; ++ni)
                    acc[mi][ni] = __builtin_amdgcn_wmma_f32_16x16x32_bf16(
                        false, af, false, bf_[ni], (short)0, acc[mi][ni],
                        false, false);
            }
        }
    }

    // ---- epilogue ----
    float* CfB = Cf ? Cf + (long long)bz * strC : nullptr;
    const float* rB = resid ? resid + (long long)bz * strR : nullptr;

#pragma unroll
    for (int mi = 0; mi < 4; ++mi) {
        int mBase = blockM + mW + 16 * mi + half * 8;  // 8 consecutive M rows
#pragma unroll
        for (int ni = 0; ni < 2; ++ni) {
            int n = blockN + nW + 16 * ni + r;
            float badd = bias ? bias[n] : 0.0f;
            float vals[8];
#pragma unroll
            for (int j = 0; j < 8; ++j) vals[j] = acc[mi][ni][j] + badd;

            if (CfB) {
#pragma unroll
                for (int j = 0; j < 8; ++j) {
                    long long idx = (long long)(mBase + j) * ldc + n;
                    float vv = vals[j];
                    if (rB) vv += rB[idx];
                    CfB[idx] = vv;
                }
            }
            if (Cb) {
                if (storeTrans) {
                    bfx8 pk;
#pragma unroll
                    for (int j = 0; j < 8; ++j) pk[j] = f2bf(vals[j]);
                    int bidx = mBase / MperBatch;
                    int mloc = mBase - bidx * MperBatch;
                    bf16* p = Cb + (long long)bidx * tbStride
                                 + (long long)n * ldct + mloc;
                    *(bfx8*)p = pk;   // contiguous along M
                } else {
                    bf16* CbB = Cb + (long long)bz * strC;
#pragma unroll
                    for (int j = 0; j < 8; ++j)
                        CbB[(long long)(mBase + j) * ldc + n] = f2bf(vals[j]);
                }
            }
        }
    }
}

// ---------------------------------------------------------------------------
// Row softmax over 1024 elements; in-place f32 + bf16 copy for the PV GEMM.
// ---------------------------------------------------------------------------
__global__ __launch_bounds__(256)
void ca_softmax_rows(float* __restrict__ S, bf16* __restrict__ P, int Mlen) {
    __shared__ float red[256];
    const long long row = blockIdx.x;
    const int tid = threadIdx.x;
    float* sr = S + row * Mlen;

    float4 v = ((const float4*)sr)[tid];
    float mx = fmaxf(fmaxf(v.x, v.y), fmaxf(v.z, v.w));
    red[tid] = mx;
    __syncthreads();
    for (int s = 128; s > 0; s >>= 1) {
        if (tid < s) red[tid] = fmaxf(red[tid], red[tid + s]);
        __syncthreads();
    }
    mx = red[0];
    __syncthreads();

    float e0 = __expf(v.x - mx), e1 = __expf(v.y - mx);
    float e2 = __expf(v.z - mx), e3 = __expf(v.w - mx);
    red[tid] = e0 + e1 + e2 + e3;
    __syncthreads();
    for (int s = 128; s > 0; s >>= 1) {
        if (tid < s) red[tid] += red[tid + s];
        __syncthreads();
    }
    float inv = 1.0f / red[0];

    float4 w;
    w.x = e0 * inv; w.y = e1 * inv; w.z = e2 * inv; w.w = e3 * inv;
    ((float4*)sr)[tid] = w;

    bf16* pr = P + row * Mlen + tid * 4;
    pr[0] = f2bf(w.x); pr[1] = f2bf(w.y); pr[2] = f2bf(w.z); pr[3] = f2bf(w.w);
}

// ---------------------------------------------------------------------------
extern "C" void kernel_launch(void* const* d_in, const int* in_sizes, int n_in,
                              void* d_out, int out_size, void* d_ws, size_t ws_size,
                              hipStream_t stream) {
    const int B = 32, N = 1024, Mc = 1024, D = 1024, DQK = 128;

    const float* x   = (const float*)d_in[0];
    const float* ctx = (const float*)d_in[1];
    const float* Wq  = (const float*)d_in[2];
    const float* bq  = (const float*)d_in[3];
    const float* Wk  = (const float*)d_in[4];
    const float* bk  = (const float*)d_in[5];
    const float* Wv  = (const float*)d_in[6];
    const float* bv  = (const float*)d_in[7];

    float* outAtt = (float*)d_out;                       // [B,N,D]
    float* outW   = outAtt + (long long)B * N * D;       // [B,N,Mc] (also scores scratch)

    // workspace carve-up (256B aligned)
    char*  ws = (char*)d_ws;
    size_t o  = 0;
    auto carve = [&](size_t bytes) {
        void* p = ws + o;
        o += (bytes + 255) & ~(size_t)255;
        return p;
    };
    bf16* xh  = (bf16*)carve((size_t)B * N * D * 2);     // x bf16
    bf16* ch  = (bf16*)carve((size_t)B * Mc * D * 2);    // context bf16
    bf16* WqT = (bf16*)carve((size_t)DQK * D * 2);       // [DQK,D]
    bf16* WkT = (bf16*)carve((size_t)DQK * D * 2);
    bf16* WvT = (bf16*)carve((size_t)D * D * 2);         // [D,D]
    bf16* Qb  = (bf16*)carve((size_t)B * N * DQK * 2);   // [B,N,DQK]
    bf16* Kb  = (bf16*)carve((size_t)B * Mc * DQK * 2);  // [B,Mc,DQK]
    bf16* VT  = (bf16*)carve((size_t)B * D * Mc * 2);    // per-batch [D,Mc]
    bf16* Pb  = (bf16*)carve((size_t)B * N * Mc * 2);    // softmax weights bf16

    // 1) casts
    {
        long long n = (long long)B * N * D;
        ca_cast_f32_bf16<<<(unsigned)(n / 1024), 256, 0, stream>>>(x, xh, n);
        ca_cast_f32_bf16<<<(unsigned)(n / 1024), 256, 0, stream>>>(ctx, ch, n);
    }
    ca_transpose_cast<<<(D * DQK + 255) / 256, 256, 0, stream>>>(Wq, WqT, D, DQK);
    ca_transpose_cast<<<(D * DQK + 255) / 256, 256, 0, stream>>>(Wk, WkT, D, DQK);
    ca_transpose_cast<<<(D * D + 255) / 256, 256, 0, stream>>>(Wv, WvT, D, D);

    // 2) Q = xh @ Wq + bq  -> bf16 [B*N, DQK]
    ca_gemm_bf16_wmma<<<dim3(DQK / BN, (B * N) / BM, 1), 256, 0, stream>>>(
        xh, 0, D, WqT, 0, D, bq,
        nullptr, Qb, 0, DQK, nullptr, 0,
        0, 0, 0, 0, B * N, DQK, D);

    // 3) K = ch @ Wk + bk  -> bf16 [B*Mc, DQK]
    ca_gemm_bf16_wmma<<<dim3(DQK / BN, (B * Mc) / BM, 1), 256, 0, stream>>>(
        ch, 0, D, WkT, 0, D, bk,
        nullptr, Kb, 0, DQK, nullptr, 0,
        0, 0, 0, 0, B * Mc, DQK, D);

    // 4) V = ch @ Wv + bv  -> bf16 stored TRANSPOSED per batch: VT[b][d][m]
    ca_gemm_bf16_wmma<<<dim3(D / BN, (B * Mc) / BM, 1), 256, 0, stream>>>(
        ch, 0, D, WvT, 0, D, bv,
        nullptr, VT, 0, D, nullptr, 0,
        1, Mc, (long long)D * Mc, Mc, B * Mc, D, D);

    // 5) scores = Q @ K^T  -> f32 into d_out weights region (batched)
    ca_gemm_bf16_wmma<<<dim3(Mc / BN, N / BM, B), 256, 0, stream>>>(
        Qb, (long long)N * DQK, DQK, Kb, (long long)Mc * DQK, DQK, nullptr,
        outW, nullptr, (long long)N * Mc, Mc, nullptr, 0,
        0, 0, 0, 0, N, Mc, DQK);

    // 6) softmax rows (in-place f32 + bf16 copy)
    ca_softmax_rows<<<B * N, 256, 0, stream>>>(outW, Pb, Mc);

    // 7) attended = P @ V + x  -> f32 into d_out (batched, residual add)
    ca_gemm_bf16_wmma<<<dim3(D / BN, N / BM, B), 256, 0, stream>>>(
        Pb, (long long)N * Mc, Mc, VT, (long long)D * Mc, Mc, nullptr,
        outAtt, nullptr, (long long)N * D, D, x, (long long)N * D,
        0, 0, 0, 0, N, D, Mc);
}
